// Head_24575802868498
// MI455X (gfx1250) — matrix operand order repaired and checked by reading, compiled, not run on hardware
//
#include <hip/hip_runtime.h>

#define B_ 4
#define T_ 4096
#define C_ 1024
#define H_ 128

typedef __attribute__((ext_vector_type(16))) __bf16 v16bf;
typedef __attribute__((ext_vector_type(8)))  float  v8f;

union FragBF {
  v16bf v;
  uint4 q[2];
  __bf16 h[16];
};

__device__ __forceinline__ v8f wmma_bf16(v16bf a, v16bf b, v8f c) {
  return __builtin_amdgcn_wmma_f32_16x16x32_bf16(
      /*neg_a=*/false, a, /*neg_b=*/false, b,
      /*c_mod=*/(short)0, c, /*reuse_a=*/false, /*reuse_b=*/false);
}

// ---------------------------------------------------------------------------
// Kernel 1: q = x@Wq, k = x@Wk, v = x@Wv   (f32 in, bf16 out)
// q,k stored [B,T,H]; v stored transposed [B,H,T] for contiguous PV B-frags.
// One block = one 16-token tile; x tile staged in LDS as bf16.
// ---------------------------------------------------------------------------
__global__ __launch_bounds__(128) void proj_qkv(
    const float* __restrict__ x,  const float* __restrict__ Wk,
    const float* __restrict__ Wq, const float* __restrict__ Wv,
    unsigned short* __restrict__ qw_, unsigned short* __restrict__ kw_,
    unsigned short* __restrict__ vtw_) {
  __shared__ __bf16 xs[16 * C_];              // 32 KB
  __bf16* qw  = (__bf16*)qw_;
  __bf16* kw  = (__bf16*)kw_;
  __bf16* vtw = (__bf16*)vtw_;

  const int tid = threadIdx.x;
  const size_t base = (size_t)blockIdx.x * 16 * C_;

  // stage 16x1024 x tile -> bf16 LDS (coalesced float4 loads, hw cvt, packed store)
  const float4* xg = (const float4*)(x + base);
  for (int i = tid; i < 16 * C_ / 4; i += 128) {
    float4 f = xg[i];
    union { ushort4 s; __bf16 h[4]; } pk;
    pk.h[0] = (__bf16)f.x; pk.h[1] = (__bf16)f.y;
    pk.h[2] = (__bf16)f.z; pk.h[3] = (__bf16)f.w;
    *((ushort4*)xs + i) = pk.s;
  }
  __syncthreads();

  const int wave = tid >> 5;
  const int l    = tid & 31;
  const int ln   = l & 15;
  const int lhi  = (l >> 4) & 1;

  // uniform matrix loop (keeps store branches wave-uniform), waves split ht tiles
  for (int mat = 0; mat < 3; ++mat) {
    const float* W = (mat == 0) ? Wq : (mat == 1) ? Wk : Wv;
    for (int ht = wave; ht < 8; ht += 4) {
      const int hc = ht * 16 + ln;            // B-frag: lane holds column N

      v8f acc = {};
#pragma unroll 1
      for (int cb = 0; cb < C_ / 32; ++cb) {
        // A frag (16x32 bf16): lane holds row M=ln; K = 8*lhi+e | 16+8*lhi+e-8
        FragBF a;
        const __bf16* ap = xs + ln * C_ + cb * 32 + 8 * lhi;
        a.q[0] = *(const uint4*)ap;
        a.q[1] = *(const uint4*)(ap + 16);
        // B frag (32x16): K = 8*lhi+e | 16+8*lhi+e-8, column hc fixed per lane
        FragBF bfrag;
        const int c0 = cb * 32 + 8 * lhi;
#pragma unroll
        for (int e = 0; e < 16; ++e)
          bfrag.h[e] = (__bf16)W[(size_t)(c0 + (e < 8 ? e : e + 8)) * H_ + hc];
        acc = wmma_bf16(a.v, bfrag.v, acc);
      }

      // store D tile: element r -> token row r+8*lhi, column hc
      if (mat == 2) {
#pragma unroll
        for (int r = 0; r < 8; ++r) {
          const int grow = (int)(blockIdx.x * 16) + r + 8 * lhi;
          const int bb = grow >> 12;          // T_ = 4096
          const int tl = grow & (T_ - 1);
          vtw[((size_t)bb * H_ + hc) * T_ + tl] = (__bf16)acc[r];
        }
      } else {
        __bf16* dst = (mat == 0) ? qw : kw;
#pragma unroll
        for (int r = 0; r < 8; ++r) {
          const int grow = (int)(blockIdx.x * 16) + r + 8 * lhi;
          dst[(size_t)grow * H_ + hc] = (__bf16)acc[r];
        }
      }
    }
  }
}

// ---------------------------------------------------------------------------
// Kernel 2: fused causal flash attention. One wave per 16-query tile.
// S = (q k^T)/32 (8 wmma per 32-key block), online softmax, O += P V (8 wmma).
// ---------------------------------------------------------------------------
__global__ __launch_bounds__(32) void flash_attn(
    const unsigned short* __restrict__ qw_, const unsigned short* __restrict__ kw_,
    const unsigned short* __restrict__ vtw_, float* __restrict__ out) {
  __shared__ __bf16 pb[16 * 32];              // P tile, C-layout -> A-layout bounce
  const __bf16* qw  = (const __bf16*)qw_;
  const __bf16* kw  = (const __bf16*)kw_;
  const __bf16* vtw = (const __bf16*)vtw_;

  const int l   = threadIdx.x & 31;
  const int ln  = l & 15;
  const int lhi = (l >> 4) & 1;
  const int qtile = blockIdx.x;               // 0..1023
  const int b   = qtile >> 8;                 // 256 tiles per batch
  const int q0  = (qtile & 255) * 16;         // local query row base
  const int q0g = qtile * 16;                 // flattened B*T row base

  // hoist Q A-fragments: 16 rows x 128 head-dim = 4 chunks of K=32
  v16bf qa[4];
#pragma unroll
  for (int dc = 0; dc < 4; ++dc) {
    FragBF f;
    const __bf16* qp = qw + (size_t)(q0g + ln) * H_ + dc * 32 + 8 * lhi;
    f.q[0] = *(const uint4*)qp;
    f.q[1] = *(const uint4*)(qp + 16);
    qa[dc] = f.v;
  }

  v8f o[8];
  float m_r[8], l_r[8];
#pragma unroll
  for (int ht = 0; ht < 8; ++ht) { v8f z = {}; o[ht] = z; }
#pragma unroll
  for (int r = 0; r < 8; ++r) { m_r[r] = -3.0e38f; l_r[r] = 0.0f; }

  const int nkb = ((q0 + 15) >> 5) + 1;       // causal: key blocks with kb0 <= imax
  const float sc = 0.03125f;                  // C^-0.5 = 1/32

  for (int kb = 0; kb < nkb; ++kb) {
    const int kb0 = kb * 32;

    // prefetch next key block's K rows and V columns into cache (global_prefetch_b8)
    if (kb + 1 < nkb) {
      const int nb0 = kb0 + 32;
      __builtin_prefetch(kw + (size_t)(b * T_ + nb0 + l) * H_, 0, 3);
      __builtin_prefetch(vtw + ((size_t)b * H_ + l * 4) * T_ + nb0, 0, 3);
    }

    // ---- S = Q K^T : two 16-key subtiles, K-loop over head dim ----
    v8f s0 = {}, s1 = {};
#pragma unroll
    for (int dc = 0; dc < 4; ++dc) {
      FragBF k0, k1;
      const size_t krow0 = (size_t)(b * T_ + kb0 + ln) * H_ + dc * 32 + 8 * lhi;
      const size_t krow1 = krow0 + (size_t)16 * H_;
      k0.q[0] = *(const uint4*)(kw + krow0);
      k0.q[1] = *(const uint4*)(kw + krow0 + 16);
      k1.q[0] = *(const uint4*)(kw + krow1);
      k1.q[1] = *(const uint4*)(kw + krow1 + 16);
      s0 = wmma_bf16(qa[dc], k0.v, s0);
      s1 = wmma_bf16(qa[dc], k1.v, s1);
    }

    // ---- causal mask + online softmax (row M=r+8*lhi lives in one 16-lane half) ----
#pragma unroll
    for (int r = 0; r < 8; ++r) {
      const int i  = q0 + r + 8 * lhi;
      const int j0 = kb0 + ln;
      const int j1 = j0 + 16;
      float a0 = (j0 <= i) ? s0[r] * sc : -3.0e38f;
      float a1 = (j1 <= i) ? s1[r] * sc : -3.0e38f;
      float rm = fmaxf(a0, a1);
#pragma unroll
      for (int msk = 1; msk <= 8; msk <<= 1)
        rm = fmaxf(rm, __shfl_xor(rm, msk, 32));
      const float mnew = fmaxf(m_r[r], rm);
      const float p0 = __expf(a0 - mnew);
      const float p1 = __expf(a1 - mnew);
      float rs = p0 + p1;
#pragma unroll
      for (int msk = 1; msk <= 8; msk <<= 1)
        rs += __shfl_xor(rs, msk, 32);
      const float corr = __expf(m_r[r] - mnew);
      m_r[r] = mnew;
      l_r[r] = l_r[r] * corr + rs;
#pragma unroll
      for (int ht = 0; ht < 8; ++ht) o[ht][r] *= corr;
      const int prow = (r + 8 * lhi) * 32;    // pb[query_row][key]
      pb[prow + ln]      = (__bf16)p0;
      pb[prow + 16 + ln] = (__bf16)p1;
    }
    __syncthreads();

    // reload P in A-matrix layout (lane holds row M=ln; K-interleave of 8)
    FragBF pa;
    const __bf16* pp = pb + ln * 32 + 8 * lhi;
    pa.q[0] = *(const uint4*)pp;
    pa.q[1] = *(const uint4*)(pp + 16);
    __syncthreads();                          // protect pb against next-iter WAR

    // ---- O += P V : vT gives contiguous key-dim loads ----
#pragma unroll
    for (int ht = 0; ht < 8; ++ht) {
      FragBF vb;
      const __bf16* vp =
          vtw + ((size_t)b * H_ + ht * 16 + ln) * T_ + kb0 + 8 * lhi;
      vb.q[0] = *(const uint4*)vp;
      vb.q[1] = *(const uint4*)(vp + 16);
      o[ht] = wmma_bf16(pa.v, vb.v, o[ht]);
    }
  }

  // ---- epilogue: divide by row sums, store f32 [B,T,H] ----
#pragma unroll
  for (int r = 0; r < 8; ++r) {
    const float inv = 1.0f / l_r[r];
    const size_t row = (size_t)(q0g + r + 8 * lhi) * H_;
#pragma unroll
    for (int ht = 0; ht < 8; ++ht)
      out[row + ht * 16 + ln] = o[ht][r] * inv;
  }
}

// ---------------------------------------------------------------------------
extern "C" void kernel_launch(void* const* d_in, const int* in_sizes, int n_in,
                              void* d_out, int out_size, void* d_ws, size_t ws_size,
                              hipStream_t stream) {
  (void)in_sizes; (void)n_in; (void)out_size; (void)ws_size;
  const float* x  = (const float*)d_in[0];
  const float* Wk = (const float*)d_in[1];
  const float* Wq = (const float*)d_in[2];
  const float* Wv = (const float*)d_in[3];

  unsigned short* qw  = (unsigned short*)d_ws;                    // [B,T,H] bf16
  unsigned short* kw  = qw + (size_t)B_ * T_ * H_;                // [B,T,H] bf16
  unsigned short* vtw = kw + (size_t)B_ * T_ * H_;                // [B,H,T] bf16

  proj_qkv<<<dim3(B_ * T_ / 16), dim3(128), 0, stream>>>(x, Wk, Wq, Wv, qw, kw, vtw);
  flash_attn<<<dim3(B_ * T_ / 16), dim3(32), 0, stream>>>(qw, kw, vtw, (float*)d_out);
}
